// Encoder_45423574122681
// MI455X (gfx1250) — compile-verified, matrix operand order checked by reading, not run on hardware
//
#include <hip/hip_runtime.h>
#include <hip/hip_bf16.h>
#include <math.h>

// ---------------- constants ----------------
#define BATCH    4
#define SEQ      1024
#define SFULL    1360          // 1024 + 256 + 64 + 16
#define DM       512
#define DFF      2048
#define NH       8
#define DK       64
#define DB       128
#define MROWS    (BATCH*SFULL) // 5440
#define SPAD     1376          // 43 * 32 (padded key dim for WMMA)
#define SCP      1377          // LDS row stride (odd -> conflict-free column reads)

typedef __attribute__((ext_vector_type(16))) __bf16 bf16x16;
typedef __attribute__((ext_vector_type(8)))  __bf16 bf16x8;
typedef __attribute__((ext_vector_type(8)))  float  f32x8;

union ABf { bf16x16 v; bf16x8 h[2]; };

__device__ __forceinline__ f32x8 wmma_bf16(const ABf& a, const ABf& b, f32x8 c) {
    return __builtin_amdgcn_wmma_f32_16x16x32_bf16(
        false, a.v, false, b.v, (short)0, c, false, false);
}

// ---------------- pyramid mask (analytic, True=allowed) ----------------
__device__ __forceinline__ void layer_of(int i, int& li, int& r) {
    if (i < 1024)      { li = 0; r = i; }
    else if (i < 1280) { li = 1; r = i - 1024; }
    else if (i < 1344) { li = 2; r = i - 1280; }
    else               { li = 3; r = i - 1344; }
}
__device__ __forceinline__ bool attn_allowed(int i, int j) {
    int li, ri, lj, rj;
    layer_of(i, li, ri); layer_of(j, lj, rj);
    if (li == lj) { int d = ri - rj; return d >= -2 && d <= 2; }
    if (li == lj + 1) return (rj >> 2) == ri;   // j is child of i
    if (lj == li + 1) return (ri >> 2) == rj;   // i is child of j
    return false;
}

// ---------------- weight pack: f32 (K,N) -> bf16 (N,K) ----------------
__global__ void pack_wT(const float* __restrict__ W, __bf16* __restrict__ out,
                        int K, int N) {
    long id = (long)blockIdx.x * 256 + threadIdx.x;
    if (id >= (long)K * N) return;
    int k = (int)(id / N), n = (int)(id % N);
    out[(long)n * K + k] = (__bf16)W[id];
}

// ---------------- embedding: circular conv(k=3) + mark linear + sinusoid PE ----
__global__ void embed_kernel(const float* __restrict__ x_enc,
                             const float* __restrict__ x_mark,
                             const float* __restrict__ tokW,   // (512,7,3)
                             const float* __restrict__ markW,  // (4,512)
                             float* __restrict__ seqF) {
    long id = (long)blockIdx.x * 256 + threadIdx.x;   // over B*1024*512
    int d = (int)(id & 511);
    int l = (int)((id >> 9) & 1023);
    int b = (int)(id >> 19);
    float acc = 0.f;
    for (int k = 0; k < 3; ++k) {
        int ls = (l + k - 1 + SEQ) & (SEQ - 1);
        const float* xr = x_enc + ((long)b * SEQ + ls) * 7;
        const float* wr = tokW + (long)d * 21 + k;
        #pragma unroll
        for (int c = 0; c < 7; ++c) acc += xr[c] * wr[c * 3];
    }
    const float* mr = x_mark + ((long)b * SEQ + l) * 4;
    #pragma unroll
    for (int m = 0; m < 4; ++m) acc += mr[m] * markW[m * DM + d];
    float expo = (float)(2 * (d / 2)) / (float)DM;
    float angle = (float)l * powf(10000.0f, -expo);
    acc += (d & 1) ? cosf(angle) : sinf(angle);
    seqF[((long)b * SFULL + l) * DM + d] = acc;
}

// ---------------- down projection: (B,1024,512)->(B,128,1024) --------------
__global__ void down_kernel(const float* __restrict__ seqF,
                            const float* __restrict__ dW,   // (512,128)
                            const float* __restrict__ db,
                            float* __restrict__ dn) {
    __shared__ float row[DM];
    int bl = blockIdx.x;               // b*1024 + l
    int b = bl >> 10, l = bl & 1023;
    int c = threadIdx.x;               // 0..127
    for (int t = c; t < DM; t += DB) row[t] = seqF[((long)b * SFULL + l) * DM + t];
    __syncthreads();
    float acc = db[c];
    for (int d = 0; d < DM; ++d) acc += row[d] * dW[d * DB + c];
    dn[((long)b * DB + c) * SEQ + l] = acc;
}

// ---------------- conv(stride4)+BatchNorm+ELU, one block per out-channel ----
__global__ void conv_bn_kernel(const float* __restrict__ in, int inStride, int inOff,
                               float* __restrict__ cat, int outOff, int Lin,
                               const float* __restrict__ cW,   // (128,128,4)
                               const float* __restrict__ cB,
                               const float* __restrict__ bg,
                               const float* __restrict__ bb) {
    __shared__ float ybuf[1024];
    __shared__ float r1[256], r2[256];
    int o = blockIdx.x, tid = threadIdx.x;
    int Lout = Lin >> 2, count = BATCH * Lout;
    for (int t = tid; t < count; t += 256) {
        int b = t / Lout, l = t % Lout;
        float acc = cB[o];
        const float* wr = cW + (long)o * DB * 4;
        for (int c = 0; c < DB; ++c) {
            const float* ir = in + ((long)b * DB + c) * inStride + inOff + l * 4;
            #pragma unroll
            for (int k = 0; k < 4; ++k) acc += ir[k] * wr[c * 4 + k];
        }
        ybuf[t] = acc;
    }
    __syncthreads();
    float s1 = 0.f, s2 = 0.f;
    for (int t = tid; t < count; t += 256) { float y = ybuf[t]; s1 += y; s2 += y * y; }
    r1[tid] = s1; r2[tid] = s2; __syncthreads();
    for (int s = 128; s > 0; s >>= 1) {
        if (tid < s) { r1[tid] += r1[tid + s]; r2[tid] += r2[tid + s]; }
        __syncthreads();
    }
    float mu = r1[0] / count;
    float var = r2[0] / count - mu * mu;
    float rs = rsqrtf(var + 1e-5f) * bg[o];
    float bet = bb[o];
    for (int t = tid; t < count; t += 256) {
        int b = t / Lout, l = t % Lout;
        float v = (ybuf[t] - mu) * rs + bet;
        v = v > 0.f ? v : (expf(v) - 1.f);               // ELU
        cat[((long)b * DB + o) * 336 + outOff + l] = v;
    }
}

// ---------------- up-proj + concat + LayerNorm (in-place on seqF) ----------
__global__ void bott_kernel(const float* __restrict__ cat,
                            const float* __restrict__ upW,  // (128,512)
                            const float* __restrict__ upB,
                            const float* __restrict__ lg,
                            const float* __restrict__ lb,
                            float* __restrict__ seqF,
                            __bf16* __restrict__ seqB) {
    __shared__ float col[DB];
    __shared__ float red[256];
    int bs = blockIdx.x;
    int b = bs / SFULL, s = bs % SFULL, tid = threadIdx.x;
    long row = (long)b * SFULL + s;
    float x0, x1;
    if (s < SEQ) {
        x0 = seqF[row * DM + tid];
        x1 = seqF[row * DM + tid + 256];
    } else {
        int p = s - SEQ;
        if (tid < DB) col[tid] = cat[((long)b * DB + tid) * 336 + p];
        __syncthreads();
        float a0 = upB[tid], a1 = upB[tid + 256];
        for (int c = 0; c < DB; ++c) {
            float cv = col[c];
            a0 += cv * upW[c * DM + tid];
            a1 += cv * upW[c * DM + tid + 256];
        }
        x0 = a0; x1 = a1;
    }
    red[tid] = x0 + x1; __syncthreads();
    for (int t = 128; t > 0; t >>= 1) { if (tid < t) red[tid] += red[tid + t]; __syncthreads(); }
    float mu = red[0] / DM; __syncthreads();
    float d0 = x0 - mu, d1 = x1 - mu;
    red[tid] = d0 * d0 + d1 * d1; __syncthreads();
    for (int t = 128; t > 0; t >>= 1) { if (tid < t) red[tid] += red[tid + t]; __syncthreads(); }
    float rs = rsqrtf(red[0] / DM + 1e-5f);
    float f0 = d0 * rs * lg[tid] + lb[tid];
    float f1 = d1 * rs * lg[tid + 256] + lb[tid + 256];
    seqF[row * DM + tid] = f0;        seqB[row * DM + tid] = (__bf16)f0;
    seqF[row * DM + tid + 256] = f1;  seqB[row * DM + tid + 256] = (__bf16)f1;
}

// ---------------- generic bf16 WMMA GEMM: out = act(A*B^T + bias) + resid ---
// A (M,K) bf16 row-major; BT (N,K) bf16 row-major.
// Register-blocked: 2 waves/block, each wave computes a 32x64 patch
// (2 M-tiles x 4 N-tiles) -> 8 WMMAs per 12 b128 loads per K=32 step.
__global__ void wmma_gemm(const __bf16* __restrict__ A,
                          const __bf16* __restrict__ BT,
                          const float* __restrict__ bias,
                          const float* __restrict__ resid,
                          float* __restrict__ outF,
                          __bf16* __restrict__ outB,
                          int M, int N, int K, int act) {
    int lane = threadIdx.x & 31;
    int wave = threadIdx.x >> 5;           // 0..1
    int tm0 = blockIdx.y * 4 + wave * 2;   // first of two M-tiles for this wave
    int nbase = blockIdx.x * 64;           // 4 N-tiles for this wave
    int ml = lane & 15;
    int koff = (lane >> 4) * 8;
    const __bf16* ap0 = A + (long)(tm0 * 16 + ml) * K;
    const __bf16* ap1 = ap0 + (long)16 * K;
    const __bf16* bp  = BT + (long)(nbase + ml) * K;
    f32x8 c0[4] = {{}, {}, {}, {}};
    f32x8 c1[4] = {{}, {}, {}, {}};
    for (int k0 = 0; k0 < K; k0 += 32) {
        ABf a0, a1;
        a0.h[0] = *(const bf16x8*)(ap0 + k0 + koff);
        a0.h[1] = *(const bf16x8*)(ap0 + k0 + 16 + koff);
        a1.h[0] = *(const bf16x8*)(ap1 + k0 + koff);
        a1.h[1] = *(const bf16x8*)(ap1 + k0 + 16 + koff);
        #pragma unroll
        for (int nt = 0; nt < 4; ++nt) {
            const __bf16* bpt = bp + (long)nt * 16 * K + k0;
            ABf bm;
            bm.h[0] = *(const bf16x8*)(bpt + koff);
            bm.h[1] = *(const bf16x8*)(bpt + 16 + koff);
            c0[nt] = wmma_bf16(a0, bm, c0[nt]);
            c1[nt] = wmma_bf16(a1, bm, c1[nt]);
        }
    }
    int rb = (lane >> 4) * 8;
    #pragma unroll
    for (int half = 0; half < 2; ++half) {
        const f32x8* cc = half ? c1 : c0;
        int rbase = (tm0 + half) * 16 + rb;
        #pragma unroll
        for (int nt = 0; nt < 4; ++nt) {
            int col = nbase + nt * 16 + ml;
            float bv = bias ? bias[col] : 0.f;
            #pragma unroll
            for (int i = 0; i < 8; ++i) {
                long row = rbase + i;
                float v = cc[nt][i] + bv;
                if (act == 1) v = 0.5f * v * (1.0f + erff(v * 0.70710678f)); // GELU
                if (resid) v += resid[row * N + col];
                if (outF) outF[row * N + col] = v;
                if (outB) outB[row * N + col] = (__bf16)v;
            }
        }
    }
}

// ---------------- V transpose: (b,s,h*64+d) -> (b,h,d, key_padded) ----------
__global__ void vtrans_kernel(const __bf16* __restrict__ V, __bf16* __restrict__ VT) {
    long id = (long)blockIdx.x * 256 + threadIdx.x;  // over B*8*64*SPAD
    long total = (long)BATCH * NH * DK * SPAD;
    if (id >= total) return;
    int s = (int)(id % SPAD);
    long r = id / SPAD;
    int d = (int)(r & 63);
    int h = (int)((r >> 6) & 7);
    int b = (int)(r >> 9);
    VT[id] = (s < SFULL) ? V[((long)b * SFULL + s) * DM + h * DK + d] : (__bf16)0.f;
}

// ---------------- fused masked attention: one wave per (b,h,16-row tile) ----
__global__ void attn_kernel(const __bf16* __restrict__ Q,
                            const __bf16* __restrict__ Kb,
                            const __bf16* __restrict__ VT,
                            __bf16* __restrict__ ctx) {
    extern __shared__ float sc[];                    // [16][SCP]
    int lane = threadIdx.x;
    int t = blockIdx.x, h = blockIdx.y, b = blockIdx.z;
    int ml = lane & 15;
    int koff = (lane >> 4) * 8;
    // Q fragments (16x64 = two K=32 steps), loaded once
    ABf aq0, aq1;
    {
        const __bf16* qp = Q + ((long)(b * SFULL + t * 16 + ml)) * DM + h * DK;
        aq0.h[0] = *(const bf16x8*)(qp + koff);
        aq0.h[1] = *(const bf16x8*)(qp + 16 + koff);
        aq1.h[0] = *(const bf16x8*)(qp + 32 + koff);
        aq1.h[1] = *(const bf16x8*)(qp + 48 + koff);
    }
    // ---- pass 1: scores = (Q K^T) / 8 with analytic pyramid mask ----
    for (int kt = 0; kt < 85; ++kt) {
        int key = kt * 16 + ml;
        const __bf16* kp = Kb + ((long)(b * SFULL + key)) * DM + h * DK;
        ABf b0, b1;
        b0.h[0] = *(const bf16x8*)(kp + koff);
        b0.h[1] = *(const bf16x8*)(kp + 16 + koff);
        b1.h[0] = *(const bf16x8*)(kp + 32 + koff);
        b1.h[1] = *(const bf16x8*)(kp + 48 + koff);
        f32x8 c = {};
        c = wmma_bf16(aq0, b0, c);
        c = wmma_bf16(aq1, b1, c);
        int col = kt * 16 + ml;
        int rb = (lane >> 4) * 8;
        #pragma unroll
        for (int i = 0; i < 8; ++i) {
            int r = rb + i;
            float v = c[i] * 0.125f;                 // 1/sqrt(64)
            sc[r * SCP + col] = attn_allowed(t * 16 + r, col) ? v : -INFINITY;
        }
    }
    __syncthreads();
    // zero-pad key columns [1360,1376)
    for (int z = lane; z < 16 * 16; z += 32)
        sc[(z >> 4) * SCP + SFULL + (z & 15)] = 0.f;
    __syncthreads();
    // ---- pass 2: row softmax (2 lanes per row, 680 cols each) ----
    {
        int r = lane & 15;
        int j0 = (lane >> 4) * 680, j1 = j0 + 680;
        float mx = -INFINITY;
        for (int j = j0; j < j1; ++j) mx = fmaxf(mx, sc[r * SCP + j]);
        mx = fmaxf(mx, __shfl_xor(mx, 16, 32));
        float sum = 0.f;
        for (int j = j0; j < j1; ++j) {
            float p = expf(sc[r * SCP + j] - mx);
            sc[r * SCP + j] = p;
            sum += p;
        }
        sum += __shfl_xor(sum, 16, 32);
        float inv = 1.0f / sum;
        for (int j = j0; j < j1; ++j) sc[r * SCP + j] *= inv;
    }
    __syncthreads();
    // ---- pass 3: ctx = P V  (K-dim = 1376 keys, 43 WMMA steps, 4 N-tiles) ----
    f32x8 acc[4] = {{}, {}, {}, {}};
    for (int ks = 0; ks < 43; ++ks) {
        int base = ks * 32;
        ABf pa;
        #pragma unroll
        for (int e = 0; e < 8; ++e) {
            pa.v[e]     = (__bf16)sc[ml * SCP + base + koff + e];
            pa.v[e + 8] = (__bf16)sc[ml * SCP + base + 16 + koff + e];
        }
        #pragma unroll
        for (int nt = 0; nt < 4; ++nt) {
            const __bf16* vp = VT + (((long)(b * NH + h) * DK) + nt * 16 + ml) * SPAD
                                  + base + koff;
            ABf vb;
            vb.h[0] = *(const bf16x8*)(vp);
            vb.h[1] = *(const bf16x8*)(vp + 16);
            acc[nt] = wmma_bf16(pa, vb, acc[nt]);
        }
    }
    int rb = (lane >> 4) * 8;
    #pragma unroll
    for (int nt = 0; nt < 4; ++nt)
        #pragma unroll
        for (int i = 0; i < 8; ++i) {
            int row = t * 16 + rb + i;
            int d = h * DK + nt * 16 + ml;
            ctx[((long)(b * SFULL + row)) * DM + d] = (__bf16)acc[nt][i];
        }
}

// ---------------- LayerNorm (post-FFN): yF -> seqF(f32) + seqB(bf16) --------
__global__ void ln_kernel(const float* __restrict__ X,
                          const float* __restrict__ lg,
                          const float* __restrict__ lb,
                          float eps,
                          float* __restrict__ seqF,
                          __bf16* __restrict__ seqB) {
    __shared__ float red[256];
    long row = blockIdx.x;
    int tid = threadIdx.x;
    float x0 = X[row * DM + tid], x1 = X[row * DM + tid + 256];
    red[tid] = x0 + x1; __syncthreads();
    for (int t = 128; t > 0; t >>= 1) { if (tid < t) red[tid] += red[tid + t]; __syncthreads(); }
    float mu = red[0] / DM; __syncthreads();
    float d0 = x0 - mu, d1 = x1 - mu;
    red[tid] = d0 * d0 + d1 * d1; __syncthreads();
    for (int t = 128; t > 0; t >>= 1) { if (tid < t) red[tid] += red[tid + t]; __syncthreads(); }
    float rs = rsqrtf(red[0] / DM + eps);
    float f0 = d0 * rs * lg[tid] + lb[tid];
    float f1 = d1 * rs * lg[tid + 256] + lb[tid + 256];
    seqF[row * DM + tid] = f0;        seqB[row * DM + tid] = (__bf16)f0;
    seqF[row * DM + tid + 256] = f1;  seqB[row * DM + tid + 256] = (__bf16)f1;
}

// ---------------- final pyramid gather ----------------
__global__ void gather_kernel(const float* __restrict__ seqF, float* __restrict__ out) {
    long id = (long)blockIdx.x * 256 + threadIdx.x;  // over B*1024*2048
    int d = (int)(id & 511);
    int j = (int)((id >> 9) & 3);
    int i = (int)((id >> 11) & 1023);
    int b = (int)(id >> 21);
    int idx = (j == 0) ? i
            : (j == 1) ? 1024 + (i >> 2)
            : (j == 2) ? 1280 + (i >> 4)
            :            1344 + (i >> 6);
    out[id] = seqF[((long)b * SFULL + idx) * DM + d];
}

// ---------------- host orchestration ----------------
extern "C" void kernel_launch(void* const* d_in, const int* in_sizes, int n_in,
                              void* d_out, int out_size, void* d_ws, size_t ws_size,
                              hipStream_t stream) {
    const float* x_enc   = (const float*)d_in[0];
    const float* x_mark  = (const float*)d_in[1];
    const float* tokW    = (const float*)d_in[2];
    const float* markW   = (const float*)d_in[3];
    const float* downW   = (const float*)d_in[4];
    const float* downB   = (const float*)d_in[5];
    const float* convW   = (const float*)d_in[6];
    const float* convB   = (const float*)d_in[7];
    const float* bnG     = (const float*)d_in[8];
    const float* bnB     = (const float*)d_in[9];
    const float* upW     = (const float*)d_in[10];
    const float* upB     = (const float*)d_in[11];
    const float* bottG   = (const float*)d_in[12];
    const float* bottB   = (const float*)d_in[13];
    const float* Wq      = (const float*)d_in[14];
    const float* bq      = (const float*)d_in[15];
    const float* Wk      = (const float*)d_in[16];
    const float* bk      = (const float*)d_in[17];
    const float* Wv      = (const float*)d_in[18];
    const float* bv      = (const float*)d_in[19];
    const float* Wo      = (const float*)d_in[20];
    const float* bo      = (const float*)d_in[21];
    const float* lnG     = (const float*)d_in[22];
    const float* lnB     = (const float*)d_in[23];
    const float* W1      = (const float*)d_in[24];
    const float* b1      = (const float*)d_in[25];
    const float* W2      = (const float*)d_in[26];
    const float* b2      = (const float*)d_in[27];
    float* out = (float*)d_out;

    // workspace carve
    char* p = (char*)d_ws;
    auto carve = [&](size_t bytes) -> void* {
        void* r = (void*)p;
        p += (bytes + 255) & ~(size_t)255;
        return r;
    };
    float*  seqF = (float*)carve((size_t)MROWS * DM * 4);
    __bf16* seqB = (__bf16*)carve((size_t)MROWS * DM * 2);
    float*  dn   = (float*)carve((size_t)BATCH * DB * SEQ * 4);
    float*  cat  = (float*)carve((size_t)BATCH * DB * 336 * 4);
    __bf16* qB   = (__bf16*)carve((size_t)MROWS * DM * 2);
    __bf16* kB   = (__bf16*)carve((size_t)MROWS * DM * 2);
    __bf16* vB   = (__bf16*)carve((size_t)MROWS * DM * 2);
    __bf16* vtB  = (__bf16*)carve((size_t)BATCH * NH * DK * SPAD * 2);
    __bf16* ctxB = (__bf16*)carve((size_t)MROWS * DM * 2);
    float*  xF   = (float*)carve((size_t)MROWS * DM * 4);
    __bf16* xB   = (__bf16*)carve((size_t)MROWS * DM * 2);
    __bf16* hB   = (__bf16*)carve((size_t)MROWS * DFF * 2);
    float*  yF   = (float*)carve((size_t)MROWS * DM * 4);
    __bf16* WqT[2], *WkT[2], *WvT[2], *WoT[2], *W1T[2], *W2T[2];
    for (int l = 0; l < 2; ++l) {
        WqT[l] = (__bf16*)carve((size_t)DM * DM * 2);
        WkT[l] = (__bf16*)carve((size_t)DM * DM * 2);
        WvT[l] = (__bf16*)carve((size_t)DM * DM * 2);
        WoT[l] = (__bf16*)carve((size_t)DM * DM * 2);
        W1T[l] = (__bf16*)carve((size_t)DM * DFF * 2);
        W2T[l] = (__bf16*)carve((size_t)DFF * DM * 2);
    }

    // pack weights -> bf16 transposed (N,K)
    int nsq = (DM * DM + 255) / 256, nff = (DM * DFF + 255) / 256;
    for (int l = 0; l < 2; ++l) {
        pack_wT<<<nsq, 256, 0, stream>>>(Wq + (long)l * DM * DM, WqT[l], DM, DM);
        pack_wT<<<nsq, 256, 0, stream>>>(Wk + (long)l * DM * DM, WkT[l], DM, DM);
        pack_wT<<<nsq, 256, 0, stream>>>(Wv + (long)l * DM * DM, WvT[l], DM, DM);
        pack_wT<<<nsq, 256, 0, stream>>>(Wo + (long)l * DM * DM, WoT[l], DM, DM);
        pack_wT<<<nff, 256, 0, stream>>>(W1 + (long)l * DM * DFF, W1T[l], DM, DFF);
        pack_wT<<<nff, 256, 0, stream>>>(W2 + (long)l * DFF * DM, W2T[l], DFF, DM);
    }

    // embedding + bottleneck
    embed_kernel<<<(BATCH * SEQ * DM) / 256, 256, 0, stream>>>(
        x_enc, x_mark, tokW, markW, seqF);
    down_kernel<<<BATCH * SEQ, DB, 0, stream>>>(seqF, downW, downB, dn);
    conv_bn_kernel<<<DB, 256, 0, stream>>>(dn, SEQ, 0, cat, 0, 1024,
        convW + 0 * DB * DB * 4, convB + 0 * DB, bnG + 0 * DB, bnB + 0 * DB);
    conv_bn_kernel<<<DB, 256, 0, stream>>>(cat, 336, 0, cat, 256, 256,
        convW + 1 * DB * DB * 4, convB + 1 * DB, bnG + 1 * DB, bnB + 1 * DB);
    conv_bn_kernel<<<DB, 256, 0, stream>>>(cat, 336, 256, cat, 320, 64,
        convW + 2 * DB * DB * 4, convB + 2 * DB, bnG + 2 * DB, bnB + 2 * DB);
    bott_kernel<<<BATCH * SFULL, 256, 0, stream>>>(cat, upW, upB, bottG, bottB,
                                                   seqF, seqB);

    // encoder layers
    // wmma_gemm: block = 64 threads (2 waves), covers 64 rows x 64 cols
    dim3 gSq(DM / 64, MROWS / 64);   // (8, 85)
    dim3 gFf(DFF / 64, MROWS / 64);  // (32, 85)
    long vtTotal = (long)BATCH * NH * DK * SPAD;
    for (int l = 0; l < 2; ++l) {
        wmma_gemm<<<gSq, 64, 0, stream>>>(seqB, WqT[l], bq + l * DM, nullptr,
                                          nullptr, qB, MROWS, DM, DM, 0);
        wmma_gemm<<<gSq, 64, 0, stream>>>(seqB, WkT[l], bk + l * DM, nullptr,
                                          nullptr, kB, MROWS, DM, DM, 0);
        wmma_gemm<<<gSq, 64, 0, stream>>>(seqB, WvT[l], bv + l * DM, nullptr,
                                          nullptr, vB, MROWS, DM, DM, 0);
        vtrans_kernel<<<(int)((vtTotal + 255) / 256), 256, 0, stream>>>(vB, vtB);
        attn_kernel<<<dim3(SFULL / 16, NH, BATCH), 32, 16 * SCP * 4, stream>>>(
            qB, kB, vtB, ctxB);
        wmma_gemm<<<gSq, 64, 0, stream>>>(ctxB, WoT[l], bo + l * DM, nullptr,
                                          xF, xB, MROWS, DM, DM, 0);
        wmma_gemm<<<gFf, 64, 0, stream>>>(xB, W1T[l], b1 + l * DFF, nullptr,
                                          nullptr, hB, MROWS, DFF, DM, 1);
        wmma_gemm<<<gSq, 64, 0, stream>>>(hB, W2T[l], b2 + l * DM, xF,
                                          yF, nullptr, MROWS, DM, DFF, 0);
        ln_kernel<<<MROWS, 256, 0, stream>>>(yF, lnG + l * DM, lnB + l * DM,
                                             1e-6f, seqF, seqB);
    }

    // gather pyramid nodes
    gather_kernel<<<(BATCH * SEQ * 4 * DM) / 256, 256, 0, stream>>>(seqF, out);
}
